// Model_37950331027526
// MI455X (gfx1250) — matrix operand order repaired and checked by reading, not compiled, not run on hardware
//
#include <hip/hip_runtime.h>
#include <hip/hip_bf16.h>
#include <math.h>

// ---------------------------------------------------------------------------
// CDNA5 / gfx1250 implementation of the hyperbolic entity-typing forward pass.
// All GEMMs use v_wmma_f32_16x16x32_bf16 (bf16 A/B, f32 accumulate), one wave
// per 32x32 C macro-tile (4 WMMAs per K step), b128 unpredicated loads in the
// steady state. Geometry (Mobius ops) is fused into block-per-row kernels
// with wave32 shuffle + LDS reductions.
// ---------------------------------------------------------------------------

typedef __attribute__((ext_vector_type(16))) __bf16 v16bf;
typedef __attribute__((ext_vector_type(8)))  float  v8f;
typedef __attribute__((ext_vector_type(4)))  float  v4f;

#define TPB 256
#define NL  5           // max elements per thread for rows up to 1280
#define EPSF 1e-15f
#define MXB (1.0f - 1e-5f)

__device__ inline __bf16 f2bf(float f) {
  unsigned u = __builtin_bit_cast(unsigned, f);
  unsigned r = u + 0x7FFFu + ((u >> 16) & 1u);   // round-to-nearest-even
  return __builtin_bit_cast(__bf16, (unsigned short)(r >> 16));
}

__device__ inline float artanh_c(float x) {
  x = fminf(fmaxf(x, -1.0f + 1e-7f), 1.0f - 1e-7f);
  return atanhf(x);
}

// ------------------------- block reductions (wave32) ------------------------
__device__ inline float blk_sum(float v) {
  __shared__ float sb[9];
  __syncthreads();
#pragma unroll
  for (int off = 16; off; off >>= 1) v += __shfl_down(v, off, 32);
  int lane = threadIdx.x & 31, wid = threadIdx.x >> 5;
  if (lane == 0) sb[wid] = v;
  __syncthreads();
  if (threadIdx.x == 0) {
    float r = 0.f;
    int nw = blockDim.x >> 5;
    for (int i = 0; i < nw; ++i) r += sb[i];
    sb[8] = r;
  }
  __syncthreads();
  return sb[8];
}

__device__ inline float blk_max(float v) {
  __shared__ float sm[9];
  __syncthreads();
#pragma unroll
  for (int off = 16; off; off >>= 1) v = fmaxf(v, __shfl_down(v, off, 32));
  int lane = threadIdx.x & 31, wid = threadIdx.x >> 5;
  if (lane == 0) sm[wid] = v;
  __syncthreads();
  if (threadIdx.x == 0) {
    float r = -3.4e38f;
    int nw = blockDim.x >> 5;
    for (int i = 0; i < nw; ++i) r = fmaxf(r, sm[i]);
    sm[8] = r;
  }
  __syncthreads();
  return sm[8];
}

// ----------------------- local-row helpers (strided) ------------------------
__device__ inline void load_local(const float* __restrict__ row, int dim,
                                  float* v, int nloc) {
  for (int i = 0; i < nloc; ++i) {
    int idx = threadIdx.x + i * TPB;
    v[i] = (idx < dim) ? row[idx] : 0.f;
  }
}
__device__ inline void store_local(float* __restrict__ row, int dim,
                                   const float* v, int nloc) {
  for (int i = 0; i < nloc; ++i) {
    int idx = threadIdx.x + i * TPB;
    if (idx < dim) row[idx] = v[i];
  }
}
__device__ inline float local_sqsum(const float* v, int nloc) {
  float s = 0.f;
  for (int i = 0; i < nloc; ++i) s += v[i] * v[i];
  return blk_sum(s);
}
__device__ inline float dev_nrm(const float* v, int nloc) {
  return sqrtf(fmaxf(local_sqsum(v, nloc), EPSF));
}
__device__ inline void dev_project(float* v, int nloc) {
  float n = dev_nrm(v, nloc);
  if (n > MXB) {
    float s = MXB / n;
    for (int i = 0; i < nloc; ++i) v[i] *= s;
  }
}
// z = mobius_add(x, y); z must not alias x or y.
__device__ inline void dev_madd(float* z, const float* x, const float* y, int nloc) {
  float sx = 0.f, sy = 0.f, sxy = 0.f;
  for (int i = 0; i < nloc; ++i) { sx += x[i]*x[i]; sy += y[i]*y[i]; sxy += x[i]*y[i]; }
  sx = blk_sum(sx); sy = blk_sum(sy); sxy = blk_sum(sxy);
  float den = fmaxf(1.f + 2.f*sxy + sx*sy, EPSF);
  float c1 = (1.f + 2.f*sxy + sy) / den;
  float c2 = (1.f - sx) / den;
  for (int i = 0; i < nloc; ++i) z[i] = c1*x[i] + c2*y[i];
}
// mv (raw x@W) -> project(tanh(mxn/xn*artanh(xn)) * mv/mxn)   (in place)
__device__ inline void dev_mv_post(float* mv, int nloc, float xn) {
  float mxn = dev_nrm(mv, nloc);
  float t = tanhf(mxn / xn * artanh_c(xn));
  float s = t / mxn;
  if (t > MXB) s *= MXB / t;
  for (int i = 0; i < nloc; ++i) mv[i] *= s;
}

// ------------------------------- WMMA GEMM ---------------------------------
__device__ inline v4f ld4(const float* p) { return *(const v4f*)p; }

__device__ inline v16bf cvt16(v4f x0, v4f x1, v4f x2, v4f x3) {
  v16bf f;
#pragma unroll
  for (int e = 0; e < 4; ++e) {
    f[e]      = f2bf(x0[e]);
    f[4 + e]  = f2bf(x1[e]);
    f[8 + e]  = f2bf(x2[e]);
    f[12 + e] = f2bf(x3[e]);
  }
  return f;
}

// C[M,N] = A[M,K] @ B.  TRANSB=0: B row-major [K,N]. TRANSB=1: B row-major [N,K].
// Requirements (met by all call sites): M%32==0, N%32==0, rows 16B-aligned.
// One wave per 32x32 C macro-tile; K main loop unguarded b128 loads; K tail guarded.
template <int TRANSB>
__global__ __launch_bounds__(256) void k_wmma_gemm(
    const float* __restrict__ A, const float* __restrict__ B,
    float* __restrict__ C, int M, int N, int K, int lda, int ldb, int ldc) {
  int tilesN = N >> 5;
  int tiles  = (M >> 5) * tilesN;
  int wave = blockIdx.x * (blockDim.x >> 5) + (threadIdx.x >> 5);
  if (wave >= tiles) return;                 // wave-uniform: EXEC all-1 for WMMA
  int tm = wave / tilesN, tn = wave % tilesN;
  int lane = threadIdx.x & 31;
  int half_sel = lane >> 4;                  // 0: lanes 0-15, 1: lanes 16-31
  int idx16 = lane & 15;
  int kabase = half_sel ? 16 : 0;            // A: lanes 0-15 K=0..15, lanes 16-31 K=16..31
  int kbbase = half_sel ? 16 : 0;            // B: lanes 0-15 K=0..15, lanes 16-31 K=16..31
  const float* arow0 = A + (size_t)(tm * 32 + idx16) * lda;
  const float* arow1 = arow0 + (size_t)16 * lda;
  int bcol0 = tn * 32 + idx16;
  int bcol1 = bcol0 + 16;
  v8f acc00 = {0.f,0.f,0.f,0.f,0.f,0.f,0.f,0.f};
  v8f acc01 = acc00, acc10 = acc00, acc11 = acc00;

  int Km = K & ~31;
  int k0 = 0;
  for (; k0 < Km; k0 += 32) {
    if (k0 + 32 < K) {                       // hint next K tile into cache
      __builtin_prefetch(arow0 + k0 + 32, 0, 1);
      __builtin_prefetch(arow1 + k0 + 32, 0, 1);
    }
    const float* r0 = arow0 + k0;
    const float* r1 = arow1 + k0;
    v16bf a0 = cvt16(ld4(r0 + kabase), ld4(r0 + kabase + 4),
                     ld4(r0 + 8 + kabase), ld4(r0 + 12 + kabase));
    v16bf a1 = cvt16(ld4(r1 + kabase), ld4(r1 + kabase + 4),
                     ld4(r1 + 8 + kabase), ld4(r1 + 12 + kabase));
    v16bf b0, b1;
    if (TRANSB) {
      const float* p0 = B + (size_t)bcol0 * ldb + k0 + kbbase;
      const float* p1 = B + (size_t)bcol1 * ldb + k0 + kbbase;
      b0 = cvt16(ld4(p0), ld4(p0 + 4), ld4(p0 + 8), ld4(p0 + 12));
      b1 = cvt16(ld4(p1), ld4(p1 + 4), ld4(p1 + 8), ld4(p1 + 12));
    } else {
      const float* bp = B + (size_t)(k0 + kbbase) * ldb;
#pragma unroll
      for (int e = 0; e < 16; ++e) {
        b0[e] = f2bf(bp[(size_t)e * ldb + bcol0]);
        b1[e] = f2bf(bp[(size_t)e * ldb + bcol1]);
      }
    }
    acc00 = __builtin_amdgcn_wmma_f32_16x16x32_bf16(false, a0, false, b0, (short)0, acc00, false, false);
    acc01 = __builtin_amdgcn_wmma_f32_16x16x32_bf16(false, a0, false, b1, (short)0, acc01, false, false);
    acc10 = __builtin_amdgcn_wmma_f32_16x16x32_bf16(false, a1, false, b0, (short)0, acc10, false, false);
    acc11 = __builtin_amdgcn_wmma_f32_16x16x32_bf16(false, a1, false, b1, (short)0, acc11, false, false);
  }
  if (k0 < K) {                              // K tail (e.g. K=300): guarded loads
    v16bf a0, a1, b0, b1;
#pragma unroll
    for (int e = 0; e < 16; ++e) {
      int ka = k0 + kabase + e;
      a0[e] = f2bf(ka < K ? arow0[ka] : 0.f);
      a1[e] = f2bf(ka < K ? arow1[ka] : 0.f);
      int kb = k0 + kbbase + e;
      float v0 = 0.f, v1 = 0.f;
      if (kb < K) {
        if (TRANSB) {
          v0 = B[(size_t)bcol0 * ldb + kb];
          v1 = B[(size_t)bcol1 * ldb + kb];
        } else {
          v0 = B[(size_t)kb * ldb + bcol0];
          v1 = B[(size_t)kb * ldb + bcol1];
        }
      }
      b0[e] = f2bf(v0);
      b1[e] = f2bf(v1);
    }
    acc00 = __builtin_amdgcn_wmma_f32_16x16x32_bf16(false, a0, false, b0, (short)0, acc00, false, false);
    acc01 = __builtin_amdgcn_wmma_f32_16x16x32_bf16(false, a0, false, b1, (short)0, acc01, false, false);
    acc10 = __builtin_amdgcn_wmma_f32_16x16x32_bf16(false, a1, false, b0, (short)0, acc10, false, false);
    acc11 = __builtin_amdgcn_wmma_f32_16x16x32_bf16(false, a1, false, b1, (short)0, acc11, false, false);
  }
  // D layout: VGPR r -> (M = base + r + 8*half_sel, N = base + idx16)
  int crow0 = tm * 32 + (half_sel ? 8 : 0);
  int ccol0 = tn * 32 + idx16;
#pragma unroll
  for (int r = 0; r < 8; ++r) {
    C[(size_t)(crow0 + r)      * ldc + ccol0]      = acc00[r];
    C[(size_t)(crow0 + r)      * ldc + ccol0 + 16] = acc01[r];
    C[(size_t)(crow0 + 16 + r) * ldc + ccol0]      = acc10[r];
    C[(size_t)(crow0 + 16 + r) * ldc + ccol0 + 16] = acc11[r];
  }
}

// ------------------------------ misc kernels -------------------------------
__global__ void k_zero(float* p, int n) {
  int i = blockIdx.x * blockDim.x + threadIdx.x;
  if (i < n) p[i] = 0.f;
}

__global__ void k_gather(const float* __restrict__ tab, const int* __restrict__ idx,
                         float* __restrict__ out, int dim) {
  int row = blockIdx.x;
  const float* src = tab + (size_t)idx[row] * dim;
  float* dst = out + (size_t)row * dim;
  for (int t = threadIdx.x; t < dim; t += blockDim.x) dst[t] = src[t];
}

__global__ void k_gather_ctx(const float* __restrict__ lut, const int* __restrict__ ids,
                             float* __restrict__ out, int L, int E, int reverse) {
  int row = blockIdx.x;
  int b = row / L, l = row % L;
  int sl = reverse ? (L - 1 - l) : l;
  const float* src = lut + (size_t)ids[b * L + sl] * E;
  float* dst = out + (size_t)row * E;
  for (int t = threadIdx.x; t < E; t += blockDim.x) dst[t] = src[t];
}

__global__ void k_gather_pe_m(const float* __restrict__ men_pos, const int* __restrict__ mentions,
                              float* __restrict__ out, int ML, int dim) {
  int row = blockIdx.x;
  int l = row % ML;
  int pos = (mentions[row] > 0) ? (l + 1) : 0;
  const float* src = men_pos + (size_t)pos * dim;
  float* dst = out + (size_t)row * dim;
  for (int t = threadIdx.x; t < dim; t += blockDim.x) dst[t] = src[t];
}

// out = mobius_add(x, y) rowwise (out may alias x or y)
__global__ void k_madd_rows(const float* __restrict__ x, const float* __restrict__ y,
                            float* __restrict__ out, int dim) {
  int row = blockIdx.x;
  int nloc = (dim + TPB - 1) / TPB;
  float xv[NL], yv[NL], zv[NL];
  load_local(x + (size_t)row * dim, dim, xv, nloc);
  load_local(y + (size_t)row * dim, dim, yv, nloc);
  dev_madd(zv, xv, yv, nloc);
  store_local(out + (size_t)row * dim, dim, zv, nloc);
}

// in-place mobius_matvec tail over nseg segments of segdim
__global__ void k_matvec_post(const float* __restrict__ x, int din,
                              float* __restrict__ mx, int segdim, int nseg) {
  int row = blockIdx.x;
  int nx = (din + TPB - 1) / TPB, ns = (segdim + TPB - 1) / TPB;
  float xv[NL], mv[NL];
  load_local(x + (size_t)row * din, din, xv, nx);
  float xn = dev_nrm(xv, nx);
  for (int g = 0; g < nseg; ++g) {
    float* seg = mx + (size_t)row * segdim * nseg + (size_t)g * segdim;
    load_local(seg, segdim, mv, ns);
    dev_mv_post(mv, ns, xn);
    store_local(seg, segdim, mv, ns);
  }
}

// full mobius_linear tail: mx (raw) -> project(madd(matvec_post, bias)) [-> expmap0(tanh(logmap0))]
__global__ void k_mlinear_post(const float* __restrict__ x, int din,
                               float* __restrict__ mx, int dout,
                               const float* __restrict__ bias, int do_tanh) {
  int row = blockIdx.x;
  int nx = (din + TPB - 1) / TPB, nm = (dout + TPB - 1) / TPB;
  float xv[NL], mv[NL], bv[NL], rz[NL];
  load_local(x + (size_t)row * din, din, xv, nx);
  float xn = dev_nrm(xv, nx);
  load_local(mx + (size_t)row * dout, dout, mv, nm);
  dev_mv_post(mv, nm, xn);
  load_local(bias, dout, bv, nm);
  dev_madd(rz, mv, bv, nm);
  dev_project(rz, nm);
  if (do_tanh) {
    float n = dev_nrm(rz, nm);
    float f = artanh_c(n) / n;
    for (int i = 0; i < nm; ++i) rz[i] = tanhf(f * rz[i]);
    float nw = dev_nrm(rz, nm);
    float g = tanhf(nw) / nw;
    for (int i = 0; i < nm; ++i) rz[i] *= g;
    dev_project(rz, nm);
  }
  store_local(mx + (size_t)row * dout, dout, rz, nm);
}

// d[row] = 2*artanh(|mobius_add(-q, k)|)
__global__ void k_pdist(const float* __restrict__ q, const float* __restrict__ k,
                        float* __restrict__ d, int dim) {
  int row = blockIdx.x;
  int nloc = (dim + TPB - 1) / TPB;
  float qv[NL], kv[NL], zv[NL];
  load_local(q + (size_t)row * dim, dim, qv, nloc);
  for (int i = 0; i < nloc; ++i) qv[i] = -qv[i];
  load_local(k + (size_t)row * dim, dim, kv, nloc);
  dev_madd(zv, qv, kv, nloc);
  float n = dev_nrm(zv, nloc);
  if (threadIdx.x == 0) d[row] = 2.f * artanh_c(n);
}

__global__ void k_softmax_neg(const float* __restrict__ d, const float* __restrict__ beta,
                              float* __restrict__ w, int Lc) {
  int b = blockIdx.x;
  float bt = beta[0];
  float x = (threadIdx.x < Lc) ? (-bt * d[(size_t)b * Lc + threadIdx.x]) : -3.4e38f;
  float m = blk_max(x);
  float e = (threadIdx.x < Lc) ? expf(x - m) : 0.f;
  float s = blk_sum(e);
  if (threadIdx.x < Lc) w[(size_t)b * Lc + threadIdx.x] = e / s;
}

// weighted gyromidpoint via Klein model. w==nullptr -> uniform weights.
__global__ void k_midpoint(const float* __restrict__ vals, const float* __restrict__ w,
                           float* __restrict__ out, int Lc, int dim) {
  int b = blockIdx.x;
  int nloc = (dim + TPB - 1) / TPB;
  float num[NL] = {0.f, 0.f, 0.f, 0.f, 0.f};
  float den = 0.f;
  float v[NL];
  for (int l = 0; l < Lc; ++l) {
    load_local(vals + ((size_t)b * Lc + l) * dim, dim, v, nloc);
    float s = local_sqsum(v, nloc);
    float dnm = 1.f + s;
    float kl2 = 4.f * s / (dnm * dnm);
    float gamma = rsqrtf(fmaxf(1.f - kl2, EPSF));
    float wt = w ? w[(size_t)b * Lc + l] : 1.f;
    float wg = wt * gamma;
    for (int i = 0; i < nloc; ++i) num[i] += wg * (2.f * v[i] / dnm);
    den += wg;
  }
  den = fmaxf(den, EPSF);
  for (int i = 0; i < nloc; ++i) num[i] /= den;
  float sk = local_sqsum(num, nloc);
  float f = 1.f / (1.f + sqrtf(fmaxf(1.f - sk, EPSF)));
  for (int i = 0; i < nloc; ++i) num[i] *= f;
  store_local(out + (size_t)b * dim, dim, num, nloc);
}

// char RNN step: hn = project(madd(madd(matvec_post(h@W), mvx[b,t]), bias))
__global__ void k_rnn_step(const float* __restrict__ h, const float* __restrict__ mh_raw,
                           const float* __restrict__ mvx, int t, int CL,
                           const float* __restrict__ bias,
                           float* __restrict__ hnew, float* __restrict__ states, int S) {
  int b = blockIdx.x;
  int n = (S + TPB - 1) / TPB;
  float hv[NL], mv[NL], xv[NL], bv[NL], t1[NL], t2[NL];
  load_local(h + (size_t)b * S, S, hv, n);
  float hn = dev_nrm(hv, n);
  load_local(mh_raw + (size_t)b * S, S, mv, n);
  dev_mv_post(mv, n, hn);
  load_local(mvx + ((size_t)b * CL + t) * S, S, xv, n);
  dev_madd(t1, mv, xv, n);
  load_local(bias, S, bv, n);
  dev_madd(t2, t1, bv, n);
  dev_project(t2, n);
  store_local(hnew + (size_t)b * S, S, t2, n);
  store_local(states + ((size_t)b * CL + t) * S, S, t2, n);
}

// GRU gates: z = sigmoid(logmap0(one_tf_z)), r = ..., rh = mobius_pw_mul(r, h)
__global__ void k_gru_gates(const float* __restrict__ h, const float* __restrict__ hr_raw,
                            const float* __restrict__ hz_raw, const float* __restrict__ mvx,
                            int l, int L, const float* __restrict__ bias,
                            float* __restrict__ zout, float* __restrict__ rhout, int S) {
  int b = blockIdx.x;
  int n = (S + TPB - 1) / TPB;
  float hv[NL], t0[NL], t1[NL], t2[NL], rv[NL];
  load_local(h + (size_t)b * S, S, hv, n);
  float hn = dev_nrm(hv, n);
  const float* mrow = mvx + ((size_t)b * L + l) * (size_t)(3 * S);
  // ---- r gate ----
  load_local(hr_raw + (size_t)b * S, S, t0, n);
  dev_mv_post(t0, n, hn);
  load_local(mrow, S, t1, n);                 // mvx_r (seg 0)
  dev_madd(t2, t0, t1, n);
  load_local(bias, S, t1, n);                 // b_r = bias[0]
  dev_madd(t0, t2, t1, n);
  float na = dev_nrm(t0, n);
  float f = artanh_c(na) / na;
  for (int i = 0; i < n; ++i) rv[i] = 1.f / (1.f + expf(-f * t0[i]));
  // ---- z gate ----
  load_local(hz_raw + (size_t)b * S, S, t0, n);
  dev_mv_post(t0, n, hn);
  load_local(mrow + 2 * S, S, t1, n);         // mvx_z (seg 2)
  dev_madd(t2, t0, t1, n);
  load_local(bias + 2 * S, S, t1, n);         // b_z = bias[2]
  dev_madd(t0, t2, t1, n);
  na = dev_nrm(t0, n);
  f = artanh_c(na) / na;
  for (int i = 0; i < n; ++i) t2[i] = 1.f / (1.f + expf(-f * t0[i]));
  store_local(zout + (size_t)b * S, S, t2, n);
  // ---- rh = mobius_pw_mul(r, h) ----
  for (int i = 0; i < n; ++i) t0[i] = rv[i] * hv[i];
  float wxn = dev_nrm(t0, n);
  float tt = tanhf(wxn / hn * artanh_c(hn));
  float s = tt / wxn;
  if (tt > MXB) s *= MXB / tt;
  for (int i = 0; i < n; ++i) t0[i] *= s;
  store_local(rhout + (size_t)b * S, S, t0, n);
}

// GRU update: hn = madd(h, pw_mul(z, madd(-h, h_tilde)))
__global__ void k_gru_update(const float* __restrict__ h, const float* __restrict__ ht_raw,
                             const float* __restrict__ rh, const float* __restrict__ mvx,
                             int l, int L, const float* __restrict__ bias,
                             const float* __restrict__ z,
                             float* __restrict__ hnew, float* __restrict__ states,
                             int l_out, int S) {
  int b = blockIdx.x;
  int n = (S + TPB - 1) / TPB;
  float hv[NL], t0[NL], t1[NL], t2[NL], dl[NL];
  load_local(rh + (size_t)b * S, S, t0, n);
  float rhn = dev_nrm(t0, n);
  load_local(ht_raw + (size_t)b * S, S, t0, n);
  dev_mv_post(t0, n, rhn);
  load_local(mvx + ((size_t)b * L + l) * (size_t)(3 * S) + S, S, t1, n);  // mvx_h (seg 1)
  dev_madd(t2, t0, t1, n);
  load_local(bias + S, S, t1, n);             // b_h = bias[1]
  dev_madd(t0, t2, t1, n);                    // h_tilde
  load_local(h + (size_t)b * S, S, hv, n);
  for (int i = 0; i < n; ++i) t1[i] = -hv[i];
  dev_madd(dl, t1, t0, n);                    // delta
  float dn = dev_nrm(dl, n);
  load_local(z + (size_t)b * S, S, t1, n);
  for (int i = 0; i < n; ++i) t0[i] = t1[i] * dl[i];  // z * delta
  float wxn = dev_nrm(t0, n);
  float tt = tanhf(wxn / dn * artanh_c(dn));
  float s = tt / wxn;
  if (tt > MXB) s *= MXB / tt;
  for (int i = 0; i < n; ++i) t0[i] *= s;     // pw
  dev_madd(t2, hv, t0, n);                    // hn
  store_local(hnew + (size_t)b * S, S, t2, n);
  store_local(states + ((size_t)b * L + l_out) * S, S, t2, n);
}

// ctx = project(madd(madd(post(fwd@Wa), post(bkwd@Wb)), bias)); out may alias fa_raw
__global__ void k_concat(const float* __restrict__ fs, const float* __restrict__ fa_raw,
                         const float* __restrict__ bs, const float* __restrict__ fb_raw,
                         const float* __restrict__ bias, float* __restrict__ ctx,
                         int S, int S2) {
  int row = blockIdx.x;
  int n1 = (S + TPB - 1) / TPB, n2 = (S2 + TPB - 1) / TPB;
  float xv[NL], av[NL], bv[NL], t1[NL], t2[NL];
  load_local(fs + (size_t)row * S, S, xv, n1);
  float fn = dev_nrm(xv, n1);
  load_local(fa_raw + (size_t)row * S2, S2, av, n2);
  dev_mv_post(av, n2, fn);
  load_local(bs + (size_t)row * S, S, xv, n1);
  float bn = dev_nrm(xv, n1);
  load_local(fb_raw + (size_t)row * S2, S2, bv, n2);
  dev_mv_post(bv, n2, bn);
  dev_madd(t1, av, bv, n2);
  load_local(bias, S2, bv, n2);
  dev_madd(t2, t1, bv, n2);
  dev_project(t2, n2);
  store_local(ctx + (size_t)row * S2, S2, t2, n2);
}

// joint = project(madd(madd(madd(post_m, post_c), post_ch), bias))
__global__ void k_final(const float* __restrict__ mvec, const float* __restrict__ jm,
                        const float* __restrict__ cvec, const float* __restrict__ jc,
                        const float* __restrict__ chvec, const float* __restrict__ jch,
                        const float* __restrict__ bias, float* __restrict__ joint,
                        int S2, int Sc, int D) {
  int b = blockIdx.x;
  int n2 = (S2 + TPB - 1) / TPB, n1 = (Sc + TPB - 1) / TPB, nD = (D + TPB - 1) / TPB;
  float xv[NL], a[NL], c[NL], r[NL];
  load_local(mvec + (size_t)b * S2, S2, xv, n2);
  float mn = dev_nrm(xv, n2);
  load_local(jm + (size_t)b * D, D, a, nD);
  dev_mv_post(a, nD, mn);
  load_local(cvec + (size_t)b * S2, S2, xv, n2);
  float cn = dev_nrm(xv, n2);
  load_local(jc + (size_t)b * D, D, c, nD);
  dev_mv_post(c, nD, cn);
  dev_madd(r, a, c, nD);                      // joint1
  load_local(chvec + (size_t)b * Sc, Sc, xv, n1);
  float chn = dev_nrm(xv, n1);
  load_local(jch + (size_t)b * D, D, a, nD);
  dev_mv_post(a, nD, chn);
  dev_madd(c, r, a, nD);                      // joint2
  load_local(bias, D, a, nD);
  dev_madd(r, c, a, nD);
  dev_project(r, nD);
  store_local(joint + (size_t)b * D, D, r, nD);
}

// hyperbolic MLR: out[b,c] = 2|a_c| asinh(2<z,a_c> / (clip(1-|z|^2) |a_c|))
__global__ void k_mlr(const float* __restrict__ joint, const float* __restrict__ p,
                      const float* __restrict__ a, float* __restrict__ out,
                      int Cn, int D) {
  int bc = blockIdx.x;
  int b = bc / Cn, c = bc % Cn;
  int nD = (D + TPB - 1) / TPB;
  float pv[NL], jv[NL], zv[NL], av[NL];
  load_local(p + (size_t)c * D, D, pv, nD);
  for (int i = 0; i < nD; ++i) pv[i] = -pv[i];
  load_local(joint + (size_t)b * D, D, jv, nD);
  dev_madd(zv, pv, jv, nD);
  load_local(a + (size_t)c * D, D, av, nD);
  float za = 0.f, z2 = 0.f, a2 = 0.f;
  for (int i = 0; i < nD; ++i) { za += zv[i]*av[i]; z2 += zv[i]*zv[i]; a2 += av[i]*av[i]; }
  za = blk_sum(za); z2 = blk_sum(z2); a2 = blk_sum(a2);
  float an = sqrtf(fmaxf(a2, EPSF));
  float num = 2.f * za;
  float den = fmaxf(1.f - z2, EPSF) * an;
  if (threadIdx.x == 0) out[(size_t)b * Cn + c] = 2.f * an * asinhf(num / den);
}

// ------------------------------- host side ---------------------------------
extern "C" void kernel_launch(void* const* d_in, const int* in_sizes, int n_in,
                              void* d_out, int out_size, void* d_ws, size_t ws_size,
                              hipStream_t stream) {
  (void)in_sizes; (void)n_in; (void)out_size; (void)ws_size;
  const int* context       = (const int*)d_in[0];
  const int* ctx_position  = (const int*)d_in[1];
  const int* mentions      = (const int*)d_in[2];
  const int* mention_chars = (const int*)d_in[3];
  const float* word_lut = (const float*)d_in[4];
  const float* char_lut = (const float*)d_in[5];
  const float* gru_Wih[2] = {(const float*)d_in[6],  (const float*)d_in[9]};
  const float* gru_Whh[2] = {(const float*)d_in[7],  (const float*)d_in[10]};
  const float* gru_b[2]   = {(const float*)d_in[8],  (const float*)d_in[11]};
  const float* w2s_W  = (const float*)d_in[12]; const float* w2s_b  = (const float*)d_in[13];
  const float* men_pos= (const float*)d_in[14];
  const float* men_Wq = (const float*)d_in[15]; const float* men_bq = (const float*)d_in[16];
  const float* men_Wk = (const float*)d_in[17]; const float* men_bk = (const float*)d_in[18];
  const float* men_beta = (const float*)d_in[19];
  const float* char_W = (const float*)d_in[20]; const float* char_U = (const float*)d_in[21];
  const float* char_b = (const float*)d_in[22];
  const float* cc_Wa  = (const float*)d_in[23]; const float* cc_Wb  = (const float*)d_in[24];
  const float* cc_bias= (const float*)d_in[25];
  const float* ctx_pos= (const float*)d_in[26];
  const float* ctx_Wq = (const float*)d_in[27]; const float* ctx_bq = (const float*)d_in[28];
  const float* ctx_Wk = (const float*)d_in[29]; const float* ctx_bk = (const float*)d_in[30];
  const float* ctx_beta = (const float*)d_in[31];
  const float* fc_Wm  = (const float*)d_in[32]; const float* fc_Wc  = (const float*)d_in[33];
  const float* fc_Wch = (const float*)d_in[34]; const float* fc_bias= (const float*)d_in[35];
  const float* mlr_p  = (const float*)d_in[36]; const float* mlr_a  = (const float*)d_in[37];
  float* out = (float*)d_out;

  const int Bb = 256, Ll = 128, MLm = 8, CLc = 16, Ee = 300, Ss = 256, Sx2 = 512,
            Dd = 1280, Cc = 128;
  const int RM = Bb * MLm;   // 2048
  const int RC = Bb * CLc;   // 4096
  const int RX = Bb * Ll;    // 32768

  // --- workspace bump allocator (floats) ---
  float* w = (float*)d_ws;
  size_t off = 0;
  auto take = [&](size_t n) { float* p = w + off; off += n; return p; };
  float* men_emb  = take((size_t)RM * Ee);
  float* mvb      = take((size_t)RM * Sx2);
  float* pe_m     = take((size_t)RM * Sx2);       // -> attn_emb (in place)
  float* qm       = take((size_t)RM * Sx2);
  float* km       = take((size_t)RM * Sx2);
  float* dm       = take((size_t)RM);
  float* wm       = take((size_t)RM);
  float* mention_vec = take((size_t)Bb * Sx2);
  float* ch_emb   = take((size_t)RC * Ss);
  float* mvx_ch   = take((size_t)RC * Ss);
  float* hA       = take((size_t)Bb * Ss);
  float* hB       = take((size_t)Bb * Ss);
  float* tmpa     = take((size_t)Bb * Ss);        // mh_raw / hr_raw
  float* tmpb     = take((size_t)Bb * Ss);        // hz_raw
  float* tmpc     = take((size_t)Bb * Ss);        // ht_raw
  float* zb       = take((size_t)Bb * Ss);
  float* rhb      = take((size_t)Bb * Ss);
  float* char_states = take((size_t)RC * Ss);
  float* char_vec = take((size_t)Bb * Ss);
  float* emb_ctx  = take((size_t)RX * Ee);
  float* mvx      = take((size_t)RX * 3 * Ss);
  float* fwdS     = take((size_t)RX * Ss);
  float* bkwS     = take((size_t)RX * Ss);
  float* fa       = take((size_t)RX * Sx2);       // -> ctx (in place)
  float* fb       = take((size_t)RX * Sx2);
  float* pec      = take((size_t)RX * Sx2);       // -> attn_emb (in place)
  float* qc       = take((size_t)RX * Sx2);
  float* kc       = take((size_t)RX * Sx2);
  float* dc       = take((size_t)RX);
  float* wc       = take((size_t)RX);
  float* ctx_vec  = take((size_t)Bb * Sx2);
  float* jm       = take((size_t)Bb * Dd);
  float* jc_      = take((size_t)Bb * Dd);
  float* jch      = take((size_t)Bb * Dd);
  float* joint    = take((size_t)Bb * Dd);

  auto gemm = [&](const float* A, const float* Bm, float* Cm, int M, int N, int K,
                  int lda, int ldb, int ldc, bool tB) {
    int tiles = (M >> 5) * (N >> 5);          // M,N multiples of 32 at all call sites
    int blocks = (tiles + 7) / 8;
    if (tB) k_wmma_gemm<1><<<blocks, 256, 0, stream>>>(A, Bm, Cm, M, N, K, lda, ldb, ldc);
    else    k_wmma_gemm<0><<<blocks, 256, 0, stream>>>(A, Bm, Cm, M, N, K, lda, ldb, ldc);
  };

  // ============================ mention encoder ============================
  k_gather<<<RM, TPB, 0, stream>>>(word_lut, mentions, men_emb, Ee);
  gemm(men_emb, w2s_W, mvb, RM, Sx2, Ee, Ee, Sx2, Sx2, false);
  k_mlinear_post<<<RM, TPB, 0, stream>>>(men_emb, Ee, mvb, Sx2, w2s_b, 1);
  k_gather_pe_m<<<RM, TPB, 0, stream>>>(men_pos, mentions, pe_m, MLm, Sx2);
  k_madd_rows<<<RM, TPB, 0, stream>>>(mvb, pe_m, pe_m, Sx2);          // attn_emb
  gemm(pe_m, men_Wq, qm, RM, Sx2, Sx2, Sx2, Sx2, Sx2, false);
  k_mlinear_post<<<RM, TPB, 0, stream>>>(pe_m, Sx2, qm, Sx2, men_bq, 0);
  gemm(pe_m, men_Wk, km, RM, Sx2, Sx2, Sx2, Sx2, Sx2, false);
  k_mlinear_post<<<RM, TPB, 0, stream>>>(pe_m, Sx2, km, Sx2, men_bk, 0);
  k_pdist<<<RM, TPB, 0, stream>>>(qm, km, dm, Sx2);
  k_softmax_neg<<<Bb, TPB, 0, stream>>>(dm, men_beta, wm, MLm);
  k_midpoint<<<Bb, TPB, 0, stream>>>(mvb, wm, mention_vec, MLm, Sx2);

  // ============================== char RNN ================================
  k_gather<<<RC, TPB, 0, stream>>>(char_lut, mention_chars, ch_emb, Ss);
  gemm(ch_emb, char_U, mvx_ch, RC, Ss, Ss, Ss, Ss, Ss, false);
  k_matvec_post<<<RC, TPB, 0, stream>>>(ch_emb, Ss, mvx_ch, Ss, 1);
  k_zero<<<(Bb * Ss + TPB - 1) / TPB, TPB, 0, stream>>>(hA, Bb * Ss);
  {
    float* hcur = hA; float* hnxt = hB;
    for (int t = 0; t < CLc; ++t) {
      gemm(hcur, char_W, tmpa, Bb, Ss, Ss, Ss, Ss, Ss, false);
      k_rnn_step<<<Bb, TPB, 0, stream>>>(hcur, tmpa, mvx_ch, t, CLc, char_b,
                                         hnxt, char_states, Ss);
      float* sw = hcur; hcur = hnxt; hnxt = sw;
    }
  }
  k_midpoint<<<Bb, TPB, 0, stream>>>(char_states, nullptr, char_vec, CLc, Ss);

  // ===================== bidirectional hyperbolic GRU ======================
  for (int dir = 0; dir < 2; ++dir) {
    float* states = (dir == 0) ? fwdS : bkwS;
    k_gather_ctx<<<RX, TPB, 0, stream>>>(word_lut, context, emb_ctx, Ll, Ee, dir);
    // input projections for all timesteps in one big WMMA GEMM: emb @ Wih^T
    gemm(emb_ctx, gru_Wih[dir], mvx, RX, 3 * Ss, Ee, Ee, Ee, 3 * Ss, true);
    k_matvec_post<<<RX, TPB, 0, stream>>>(emb_ctx, Ee, mvx, Ss, 3);
    k_zero<<<(Bb * Ss + TPB - 1) / TPB, TPB, 0, stream>>>(hA, Bb * Ss);
    float* hcur = hA; float* hnxt = hB;
    const float* Whh = gru_Whh[dir];
    const float* bia = gru_b[dir];
    for (int l = 0; l < Ll; ++l) {
      gemm(hcur, Whh,                       tmpa, Bb, Ss, Ss, Ss, Ss, Ss, true); // W_hr
      gemm(hcur, Whh + (size_t)2 * Ss * Ss, tmpb, Bb, Ss, Ss, Ss, Ss, Ss, true); // W_hz
      k_gru_gates<<<Bb, TPB, 0, stream>>>(hcur, tmpa, tmpb, mvx, l, Ll, bia, zb, rhb, Ss);
      gemm(rhb, Whh + (size_t)Ss * Ss,      tmpc, Bb, Ss, Ss, Ss, Ss, Ss, true); // W_hhh
      int l_out = (dir == 0) ? l : (Ll - 1 - l);
      k_gru_update<<<Bb, TPB, 0, stream>>>(hcur, tmpc, rhb, mvx, l, Ll, bia, zb,
                                           hnxt, states, l_out, Ss);
      float* sw = hcur; hcur = hnxt; hnxt = sw;
    }
  }

  // ===================== context concat + attention ========================
  gemm(fwdS, cc_Wa, fa, RX, Sx2, Ss, Ss, Sx2, Sx2, false);
  gemm(bkwS, cc_Wb, fb, RX, Sx2, Ss, Ss, Sx2, Sx2, false);
  k_concat<<<RX, TPB, 0, stream>>>(fwdS, fa, bkwS, fb, cc_bias, fa, Ss, Sx2); // ctx = fa
  k_gather<<<RX, TPB, 0, stream>>>(ctx_pos, ctx_position, pec, Sx2);
  k_madd_rows<<<RX, TPB, 0, stream>>>(fa, pec, pec, Sx2);                     // attn_emb
  gemm(pec, ctx_Wq, qc, RX, Sx2, Sx2, Sx2, Sx2, Sx2, false);
  k_mlinear_post<<<RX, TPB, 0, stream>>>(pec, Sx2, qc, Sx2, ctx_bq, 0);
  gemm(pec, ctx_Wk, kc, RX, Sx2, Sx2, Sx2, Sx2, Sx2, false);
  k_mlinear_post<<<RX, TPB, 0, stream>>>(pec, Sx2, kc, Sx2, ctx_bk, 0);
  k_pdist<<<RX, TPB, 0, stream>>>(qc, kc, dc, Sx2);
  k_softmax_neg<<<Bb, TPB, 0, stream>>>(dc, ctx_beta, wc, Ll);
  k_midpoint<<<Bb, TPB, 0, stream>>>(fa, wc, ctx_vec, Ll, Sx2);

  // ========================= full concat + MLR =============================
  gemm(mention_vec, fc_Wm,  jm,  Bb, Dd, Sx2, Sx2, Dd, Dd, false);
  gemm(ctx_vec,     fc_Wc,  jc_, Bb, Dd, Sx2, Sx2, Dd, Dd, false);
  gemm(char_vec,    fc_Wch, jch, Bb, Dd, Ss,  Ss,  Dd, Dd, false);
  k_final<<<Bb, TPB, 0, stream>>>(mention_vec, jm, ctx_vec, jc_, char_vec, jch,
                                  fc_bias, joint, Sx2, Ss, Dd);
  k_mlr<<<Bb * Cc, TPB, 0, stream>>>(joint, mlr_p, mlr_a, out, Cc, Dd);
}
